// DiffAttention_66279935311921
// MI455X (gfx1250) — compile-verified, matrix operand order checked by reading, not run on hardware
//
#include <hip/hip_runtime.h>
#include <cstddef>
#include <cstdint>

// ---------------- types ----------------
typedef unsigned short u16;
typedef unsigned int   u32;
typedef __attribute__((ext_vector_type(16))) __bf16 v16bf;
typedef __attribute__((ext_vector_type(8)))  float  v8f;

union Frag16 { uint4 u[2]; v16bf v; };
union V4U16  { uint4 q; u16 s[8]; };

#if __has_builtin(__builtin_amdgcn_permlane16)
#define HAVE_PERMLANE16 1
#else
#define HAVE_PERMLANE16 0
#endif

#if __has_builtin(__builtin_amdgcn_global_load_async_to_lds_b128)
#define HAVE_ASYNC_LDS 1
typedef int v4i_t __attribute__((ext_vector_type(4)));
typedef __attribute__((address_space(1))) v4i_t gv4i;  // global-AS int4
typedef __attribute__((address_space(3))) v4i_t lv4i;  // LDS-AS int4
#else
#define HAVE_ASYNC_LDS 0
#endif

__device__ __forceinline__ u16 f2bf(float f) {
  u32 u = __float_as_uint(f);
  u32 r = (u + 0x7FFFu + ((u >> 16) & 1u)) >> 16;   // round-to-nearest-even
  return (u16)r;
}

// Load one 16-lane bf16 fragment: two contiguous 8-element runs at
// K = half*8 and K = 16 + half*8 (CDNA5 16-bit A/B layout, wave32).
__device__ __forceinline__ void load_frag(Frag16& f, const u16* p, int hf) {
  f.u[0] = *(const uint4*)(p + hf * 8);
  f.u[1] = *(const uint4*)(p + 16 + hf * 8);
}

__device__ __forceinline__ v8f wmma_bf16(v16bf a, v16bf b, v8f c) {
  return __builtin_amdgcn_wmma_f32_16x16x32_bf16(false, a, false, b, (short)0, c,
                                                 false, false);
}

// ---- 16-lane xor-butterfly reductions.
// Prefer v_permlane16_b32 (pure VALU, no LDS traffic / dscnt waits) over
// __shfl_xor's ds_bpermute lowering.
#if HAVE_PERMLANE16
__device__ __forceinline__ float pl16(float v, unsigned s0, unsigned s1) {
  unsigned u = __float_as_uint(v);
  unsigned r = __builtin_amdgcn_permlane16(u, u, s0, s1, false, false);
  return __uint_as_float(r);
}
__device__ __forceinline__ float redmax16(float v) {
  v = fmaxf(v, pl16(v, 0x67452301u, 0xEFCDAB89u)); // xor 1
  v = fmaxf(v, pl16(v, 0x54761032u, 0xDCFE98BAu)); // xor 2
  v = fmaxf(v, pl16(v, 0x32107654u, 0xBA98FEDCu)); // xor 4
  v = fmaxf(v, pl16(v, 0xFEDCBA98u, 0x76543210u)); // xor 8
  return v;
}
__device__ __forceinline__ float redsum16(float v) {
  v += pl16(v, 0x67452301u, 0xEFCDAB89u);
  v += pl16(v, 0x54761032u, 0xDCFE98BAu);
  v += pl16(v, 0x32107654u, 0xBA98FEDCu);
  v += pl16(v, 0xFEDCBA98u, 0x76543210u);
  return v;
}
#else
__device__ __forceinline__ float redmax16(float v) {
#pragma unroll
  for (int m = 1; m < 16; m <<= 1) v = fmaxf(v, __shfl_xor(v, m, 32));
  return v;
}
__device__ __forceinline__ float redsum16(float v) {
#pragma unroll
  for (int m = 1; m < 16; m <<= 1) v += __shfl_xor(v, m, 32);
  return v;
}
#endif

// ---------------- fp32 -> bf16 convert ----------------
__global__ __launch_bounds__(256) void f32_to_bf16_k(const float* __restrict__ in,
                                                     u16* __restrict__ out, int n) {
  for (int i = blockIdx.x * blockDim.x + threadIdx.x; i < n;
       i += gridDim.x * blockDim.x)
    out[i] = f2bf(in[i]);
}

// ---------------- lambda scalar ----------------
__global__ void lambda_k(const float* __restrict__ lq1, const float* __restrict__ lk1,
                         const float* __restrict__ lq2, const float* __restrict__ lk2,
                         float* __restrict__ lam) {
  if (threadIdx.x == 0 && blockIdx.x == 0) {
    float s1 = 0.f, s2 = 0.f;
    for (int i = 0; i < 64; ++i) {
      s1 += lq1[i] * lk1[i];
      s2 += lq2[i] * lk2[i];
    }
    *lam = __expf(s1) - __expf(s2) + 0.8f;
  }
}

// ---------------- GEMM: Y = (A @ W^T + bias) * alpha ----------------
// A: [M,K] bf16 row-major.  W: [N,K] bf16 row-major.  Tile 128x64 per block,
// 8 waves x (16 rows x 64 cols), 4 accumulators per wave.
template <bool OUT_BF16>
__global__ __launch_bounds__(256) void gemm_bf16_k(
    const u16* __restrict__ A, const u16* __restrict__ W,
    const float* __restrict__ bias, void* __restrict__ out, int M, int Nn, int K,
    float alpha) {
  const int tid = threadIdx.x, wave = tid >> 5, lane = tid & 31;
  const int hf = lane >> 4, l15 = lane & 15;
  const int r0 = blockIdx.x * 128 + wave * 16;
  const int c0 = blockIdx.y * 64;

  const v8f zero = {0.f, 0.f, 0.f, 0.f, 0.f, 0.f, 0.f, 0.f};
  v8f acc[4];
#pragma unroll
  for (int nt = 0; nt < 4; ++nt) acc[nt] = zero;

  const u16* arow = A + (size_t)(r0 + l15) * K;
  const u16* wrow = W + (size_t)(c0 + l15) * K;

  for (int kk = 0; kk < K; kk += 32) {
    Frag16 a;
    load_frag(a, arow + kk, hf);
    __builtin_prefetch(arow + (kk + 32 < K ? kk + 32 : 0), 0, 0);
    __builtin_prefetch(wrow + (kk + 32 < K ? kk + 32 : 0), 0, 0);
#pragma unroll
    for (int nt = 0; nt < 4; ++nt) {
      Frag16 bf;
      load_frag(bf, wrow + (size_t)nt * 16 * K + kk, hf);
      acc[nt] = wmma_bf16(a.v, bf.v, acc[nt]);
    }
  }

#pragma unroll
  for (int nt = 0; nt < 4; ++nt) {
#pragma unroll
    for (int r = 0; r < 8; ++r) {
      const int n = c0 + nt * 16 + l15;
      const int m = r0 + r + 8 * hf;
      const float v = (acc[nt][r] + bias[n]) * alpha;
      if (OUT_BF16)
        ((u16*)out)[(size_t)m * Nn + n] = f2bf(v);
      else
        ((float*)out)[(size_t)m * Nn + n] = v;
    }
  }
}

// ---------------- Differential flash attention ----------------
// grid: (qtile=32, e=8, b=2); block: 128 threads (4 waves x 16 q rows).
// Per 32-key chunk: stage K (2 heads x 32x64) async to LDS and V-transposed
// (128x32), score WMMAs, online softmax (both heads), probs via LDS -> PV WMMAs.
__global__ __launch_bounds__(128) void diffattn_k(
    const u16* __restrict__ qbf, const u16* __restrict__ kbf,
    const u16* __restrict__ vbf, const float* __restrict__ norm_w,
    const float* __restrict__ lam_p, u16* __restrict__ attn) {
  constexpr int NN = 2048;
  const int qt = blockIdx.x, e = blockIdx.y, b = blockIdx.z;
  const int tid = threadIdx.x, wave = tid >> 5, lane = tid & 31;
  const int hf = lane >> 4, l15 = lane & 15;
  const int q0 = qt * 64 + wave * 16;

  __shared__ __align__(16) u16 lds_k[2 * 32 * 64];   // [h][key][d]
  __shared__ __align__(16) u16 lds_vt[128 * 32];     // [d][key]   (transposed)
  __shared__ __align__(16) u16 lds_p[4][2][16 * 32]; // [wave][h][row][key]

  // q fragments for both heads (d split 0..31 / 32..63), loaded once
  Frag16 qa[2][2];
  const u16* qrow = qbf + (size_t)(b * NN + q0 + l15) * 1024 + e * 128;
#pragma unroll
  for (int h = 0; h < 2; ++h)
#pragma unroll
    for (int dc = 0; dc < 2; ++dc) load_frag(qa[h][dc], qrow + h * 64 + dc * 32, hf);

  const v8f zero = {0.f, 0.f, 0.f, 0.f, 0.f, 0.f, 0.f, 0.f};
  v8f O[2][8];
  float mr[2][8], lr[2][8];
#pragma unroll
  for (int h = 0; h < 2; ++h)
#pragma unroll
    for (int r = 0; r < 8; ++r) {
      mr[h][r] = -3.0e38f;
      lr[h][r] = 0.f;
    }
#pragma unroll
  for (int h = 0; h < 2; ++h)
#pragma unroll
    for (int db = 0; db < 8; ++db) O[h][db] = zero;

  const float lam = lam_p[0];
  float nw[8];
#pragma unroll
  for (int db = 0; db < 8; ++db) nw[db] = norm_w[db * 16 + l15];

  for (int kc = 0; kc < NN; kc += 32) {
    __syncthreads();  // protect LDS K/V reuse across iterations

    // ---- stage K chunk: 2 heads x 32 keys x 64 dims; async global->LDS b128
    // (no VGPR round-trip, tracked by ASYNCcnt) when available.
    for (int u = tid; u < 512; u += 128) {
      const int off = u * 8;
      const int h = off >> 11, rem = off & 2047;
      const int k = rem >> 6, d = rem & 63;
      const u16* src =
          kbf + (size_t)(b * NN + kc + k) * 1024 + (2 * e + h) * 64 + d;
#if HAVE_ASYNC_LDS
      __builtin_amdgcn_global_load_async_to_lds_b128(
          (gv4i*)src, (lv4i*)((uint4*)lds_k + u), 0, 0);
#else
      ((uint4*)lds_k)[u] = *(const uint4*)src;
#endif
    }
    // ---- stage V chunk transposed: read rows of V, scatter columns
    for (int u = tid; u < 512; u += 128) {
      const int off = u * 8;
      const int k = off >> 7, d0 = off & 127;
      V4U16 tv;
      tv.q = *(const uint4*)(vbf + (size_t)(b * NN + kc + k) * 1024 + e * 128 + d0);
#pragma unroll
      for (int j = 0; j < 8; ++j) lds_vt[(d0 + j) * 32 + k] = tv.s[j];
    }
#if HAVE_ASYNC_LDS
    asm volatile("s_wait_asynccnt 0" ::: "memory");
#endif
    __syncthreads();

    // ---- scores + online softmax, both heads
#pragma unroll
    for (int h = 0; h < 2; ++h) {
      v8f S0 = zero, S1 = zero;
      const u16* kb = lds_k + h * 2048;
#pragma unroll
      for (int dc = 0; dc < 2; ++dc) {
        Frag16 b0, b1;
        load_frag(b0, kb + (size_t)(l15)*64 + dc * 32, hf);        // keys 0..15
        load_frag(b1, kb + (size_t)(16 + l15) * 64 + dc * 32, hf); // keys 16..31
        S0 = wmma_bf16(qa[h][dc].v, b0.v, S0);
        S1 = wmma_bf16(qa[h][dc].v, b1.v, S1);
      }
#pragma unroll
      for (int r = 0; r < 8; ++r) {
        const float cmax = redmax16(fmaxf(S0[r], S1[r]));
        const float mnew = fmaxf(mr[h][r], cmax);
        const float al = __expf(mr[h][r] - mnew);
        const float p0 = __expf(S0[r] - mnew);
        const float p1 = __expf(S1[r] - mnew);
        lr[h][r] = lr[h][r] * al + redsum16(p0 + p1);
        mr[h][r] = mnew;
#pragma unroll
        for (int db = 0; db < 8; ++db) O[h][db][r] *= al;
        const int row = r + 8 * hf;  // C-layout row owned by this lane half
        lds_p[wave][h][row * 32 + l15] = f2bf(p0);
        lds_p[wave][h][row * 32 + 16 + l15] = f2bf(p1);
      }
    }
    // per-wave LDS round-trip (C-layout -> A-layout); same-wave DS is in-order,
    // wait explicitly for the b16 stores before the b128 reads below.
    asm volatile("s_wait_dscnt 0" ::: "memory");

    Frag16 pf0, pf1;
    load_frag(pf0, &lds_p[wave][0][l15 * 32], hf);
    load_frag(pf1, &lds_p[wave][1][l15 * 32], hf);
#pragma unroll
    for (int db = 0; db < 8; ++db) {
      Frag16 vf;
      load_frag(vf, lds_vt + (size_t)(db * 16 + l15) * 32, hf);
      O[0][db] = wmma_bf16(pf0.v, vf.v, O[0][db]);
      O[1][db] = wmma_bf16(pf1.v, vf.v, O[1][db]);
    }
  }

  // ---- epilogue: diff, normalize, RMSNorm(128), norm_w, (1 - lambda_init)
  const float oscale = 0.2f;
#pragma unroll
  for (int r = 0; r < 8; ++r) {
    const float i0 = 1.0f / lr[0][r];
    const float i1 = lam / lr[1][r];
    float od[8], ss = 0.f;
#pragma unroll
    for (int db = 0; db < 8; ++db) {
      od[db] = O[0][db][r] * i0 - O[1][db][r] * i1;
      ss += od[db] * od[db];
    }
    ss = redsum16(ss);
    const float rms = rsqrtf(ss * (1.0f / 128.0f) + 1e-5f);
    const int row = b * NN + q0 + r + 8 * hf;
    u16* op = attn + (size_t)row * 1024 + e * 128;
#pragma unroll
    for (int db = 0; db < 8; ++db)
      op[db * 16 + l15] = f2bf(od[db] * rms * nw[db] * oscale);
  }
}

// ---------------- host launch ----------------
extern "C" void kernel_launch(void* const* d_in, const int* in_sizes, int n_in,
                              void* d_out, int out_size, void* d_ws, size_t ws_size,
                              hipStream_t stream) {
  (void)in_sizes; (void)n_in; (void)out_size; (void)ws_size;
  const float* x      = (const float*)d_in[0];
  const float* Wq     = (const float*)d_in[1];
  const float* bq     = (const float*)d_in[2];
  const float* Wk     = (const float*)d_in[3];
  const float* bk     = (const float*)d_in[4];
  const float* Wv     = (const float*)d_in[5];
  const float* bv     = (const float*)d_in[6];
  const float* Wo     = (const float*)d_in[7];
  const float* bo     = (const float*)d_in[8];
  const float* norm_w = (const float*)d_in[9];
  const float* lq1    = (const float*)d_in[10];
  const float* lk1    = (const float*)d_in[11];
  const float* lq2    = (const float*)d_in[12];
  const float* lk2    = (const float*)d_in[13];

  char* ws = (char*)d_ws;
  constexpr size_t MB = 1024 * 1024;
  u16* xbf   = (u16*)(ws + 0);            // 8 MB
  u16* wqbf  = (u16*)(ws + 8 * MB);       // 2 MB each
  u16* wkbf  = (u16*)(ws + 10 * MB);
  u16* wvbf  = (u16*)(ws + 12 * MB);
  u16* wobf  = (u16*)(ws + 14 * MB);
  u16* qbf   = (u16*)(ws + 16 * MB);      // 8 MB each
  u16* kbf   = (u16*)(ws + 24 * MB);
  u16* vbf   = (u16*)(ws + 32 * MB);
  u16* abf   = (u16*)(ws + 40 * MB);
  float* lam = (float*)(ws + 48 * MB);

  f32_to_bf16_k<<<512, 256, 0, stream>>>(x, xbf, 4096 * 1024);
  f32_to_bf16_k<<<256, 256, 0, stream>>>(Wq, wqbf, 1024 * 1024);
  f32_to_bf16_k<<<256, 256, 0, stream>>>(Wk, wkbf, 1024 * 1024);
  f32_to_bf16_k<<<256, 256, 0, stream>>>(Wv, wvbf, 1024 * 1024);
  f32_to_bf16_k<<<256, 256, 0, stream>>>(Wo, wobf, 1024 * 1024);
  lambda_k<<<1, 32, 0, stream>>>(lq1, lk1, lq2, lk2, lam);

  const dim3 gg(32, 16);
  // q gets the 1/sqrt(64) scale fused into the projection epilogue
  gemm_bf16_k<true><<<gg, 256, 0, stream>>>(xbf, wqbf, bq, qbf, 4096, 1024, 1024,
                                            0.125f);
  gemm_bf16_k<true><<<gg, 256, 0, stream>>>(xbf, wkbf, bk, kbf, 4096, 1024, 1024,
                                            1.0f);
  gemm_bf16_k<true><<<gg, 256, 0, stream>>>(xbf, wvbf, bv, vbf, 4096, 1024, 1024,
                                            1.0f);

  diffattn_k<<<dim3(32, 8, 2), 128, 0, stream>>>(qbf, kbf, vbf, norm_w, lam, abf);

  gemm_bf16_k<false><<<gg, 256, 0, stream>>>(abf, wobf, bo, d_out, 4096, 1024,
                                             1024, 1.0f);
}
